// model_63797444215576
// MI455X (gfx1250) — matrix operand order repaired and checked by reading, not compiled, not run on hardware
//
#include <hip/hip_runtime.h>

// ---------------------------------------------------------------------------
// GCN forward for MI455X (gfx1250, wave32, WMMA).
// B=8192, N=48 nodes, F=64 features, L=11 blocks.
// Memory-bound (~14 FLOP/B): fuse (att @ (h@W) + bias) + batch-stats into one
// WMMA pass, and BN+sigmoid+residual into one elementwise pass per layer.
// ---------------------------------------------------------------------------

typedef __attribute__((ext_vector_type(16))) _Float16 v16h;
typedef __attribute__((ext_vector_type(8)))  float    v8f;

union Frag16 { v16h v; uint4 q[2]; };

#define B_TOT   8192
#define N_NODE  48
#define F_FEAT  64
#define NF      3072          // N_NODE * F_FEAT
#define L_LAYER 11
#define BN_EPS  1e-5f
#define BATCH_PER_WG 64
#define GEMM_WGS (B_TOT / BATCH_PER_WG)   // 128

// ---------------------------------------------------------------------------
// Fused GEMM pass:  t[b] = att @ (h[b] @ W) + bias  == (att @ h[b]) @ W + bias
// Also accumulates per-feature sum / sum^2 over the batch for BatchNorm.
// One WG = 128 threads (4 waves); each wave owns 3 of the 12 output 16x16
// tiles (fixed across batch items -> stats stay in registers).
// ---------------------------------------------------------------------------
__global__ __launch_bounds__(128, 1)
void gcn_gemm_kernel(const float* __restrict__ h,
                     const float* __restrict__ Wl,     // [64][64] (f,g)
                     const float* __restrict__ attl,   // [48][48] (n,m)
                     const float* __restrict__ biasl,  // [64]
                     float* __restrict__ tout,         // [B][48][64] pre-BN
                     float* __restrict__ sumb,         // [3072]
                     float* __restrict__ sqb)          // [3072]
{
    // LDS (28 KB total, f16):
    __shared__ __align__(16) _Float16 sAtt[N_NODE * 64]; // A1: [n][m], m padded 48->64 w/ 0
    __shared__ __align__(16) _Float16 sWt [64 * 64];     // B2: [g][f]  (W transposed)
    __shared__ __align__(16) _Float16 sHT [64 * 64];     // B1: [f][m], m padded w/ 0
    __shared__ __align__(16) _Float16 sT  [N_NODE * 64]; // A2: [n][f]  (att@h, f16)

    const int tid  = threadIdx.x;
    const int lane = tid & 31;
    const int wave = tid >> 5;
    const int col  = lane & 15;   // A: row-in-tile, B: col-in-tile
    const int hi   = lane >> 4;   // K half selector

    // Stage att (zero-padded K) and W^T once per WG (tiny, L2-resident).
    for (int idx = tid; idx < N_NODE * 64; idx += 128) {
        int n = idx >> 6, m = idx & 63;
        sAtt[idx] = (_Float16)((m < N_NODE) ? attl[n * N_NODE + m] : 0.0f);
    }
    for (int idx = tid; idx < 64 * 64; idx += 128) {
        int g = idx >> 6, f = idx & 63;
        sWt[idx] = (_Float16)Wl[f * 64 + g];
    }

    // Per-tile bias (constant over accumulator rows).
    float biasv[3];
    #pragma unroll
    for (int t = 0; t < 3; ++t) {
        int cn = (wave * 3 + t) & 3;
        biasv[t] = biasl[cn * 16 + col];
    }

    // Register-resident batch statistics (feature is fixed per (t, r, lane)).
    float accS[3][8], accQ[3][8];
    #pragma unroll
    for (int t = 0; t < 3; ++t)
        #pragma unroll
        for (int r = 0; r < 8; ++r) { accS[t][r] = 0.0f; accQ[t][r] = 0.0f; }

    const int b0 = blockIdx.x * BATCH_PER_WG;
    for (int it = 0; it < BATCH_PER_WG; ++it) {
        const int b = b0 + it;

        // h[b] -> LDS transposed f16: sHT[f][m], nodes 48..63 zero-padded.
        const float* hb = h + (size_t)b * (N_NODE * F_FEAT);
        for (int idx = tid; idx < 64 * 64; idx += 128) {
            int f = idx >> 6, m = idx & 63;
            sHT[idx] = (_Float16)((m < N_NODE) ? hb[m * F_FEAT + f] : 0.0f);
        }
        __syncthreads();

        // Stage 1: T = att @ h[b]   (M=48, N=64, K=48 padded to 64)
        #pragma unroll
        for (int t = 0; t < 3; ++t) {
            int tileIdx = wave * 3 + t;
            int tm = tileIdx >> 2, cn = tileIdx & 3;
            v8f acc = {};
            #pragma unroll
            for (int kc = 0; kc < 2; ++kc) {
                Frag16 A, Bf;
                // A frag: lane row m=tm*16+col, K = kc*32 + hi*8 + {0..7, 16..23}
                const _Float16* arow = &sAtt[(tm * 16 + col) * 64 + kc * 32 + hi * 8];
                A.q[0] = *(const uint4*)(arow);
                A.q[1] = *(const uint4*)(arow + 16);
                // B frag: lane col n=cn*16+col, K = kc*32 + hi*8 + {0..7, 16..23}
                const _Float16* brow = &sHT[(cn * 16 + col) * 64 + kc * 32 + hi * 8];
                Bf.q[0] = *(const uint4*)(brow);
                Bf.q[1] = *(const uint4*)(brow + 16);
                acc = __builtin_amdgcn_wmma_f32_16x16x32_f16(false, A.v, false, Bf.v,
                                                             (short)0, acc, false, false);
            }
            // D layout: (m = r + 8*hi, n = col) -> sT row-major [n][f]
            #pragma unroll
            for (int r = 0; r < 8; ++r)
                sT[(tm * 16 + r + 8 * hi) * 64 + cn * 16 + col] = (_Float16)acc[r];
        }
        __syncthreads();

        // Stage 2: out = T @ W + bias   (M=48, N=64, K=64)
        #pragma unroll
        for (int t = 0; t < 3; ++t) {
            int tileIdx = wave * 3 + t;
            int tm = tileIdx >> 2, cn = tileIdx & 3;
            v8f acc = {};
            #pragma unroll
            for (int kc = 0; kc < 2; ++kc) {
                Frag16 A, Bf;
                const _Float16* arow = &sT[(tm * 16 + col) * 64 + kc * 32 + hi * 8];
                A.q[0] = *(const uint4*)(arow);
                A.q[1] = *(const uint4*)(arow + 16);
                const _Float16* brow = &sWt[(cn * 16 + col) * 64 + kc * 32 + hi * 8];
                Bf.q[0] = *(const uint4*)(brow);
                Bf.q[1] = *(const uint4*)(brow + 16);
                acc = __builtin_amdgcn_wmma_f32_16x16x32_f16(false, A.v, false, Bf.v,
                                                             (short)0, acc, false, false);
            }
            float* orow = tout + (size_t)b * (N_NODE * F_FEAT);
            #pragma unroll
            for (int r = 0; r < 8; ++r) {
                int node = tm * 16 + r + 8 * hi;
                int g    = cn * 16 + col;
                float v  = acc[r] + biasv[t];
                orow[node * F_FEAT + g] = v;
                accS[t][r] += v;
                accQ[t][r] += v * v;
            }
        }
        __syncthreads();   // protect sHT/sT before next iteration
    }

    // Flush per-WG stats (one atomic pair per feature per WG: ~786K/layer).
    #pragma unroll
    for (int t = 0; t < 3; ++t) {
        int tileIdx = wave * 3 + t;
        int tm = tileIdx >> 2, cn = tileIdx & 3;
        #pragma unroll
        for (int r = 0; r < 8; ++r) {
            int flat = (tm * 16 + r + 8 * hi) * F_FEAT + cn * 16 + col;
            atomicAdd(&sumb[flat], accS[t][r]);
            atomicAdd(&sqb[flat],  accQ[t][r]);
        }
    }
}

// ---------------------------------------------------------------------------
// BN (batch stats) + sigmoid + optional residual add, float4-vectorized.
// ---------------------------------------------------------------------------
__global__ __launch_bounds__(256)
void bn_sigmoid_kernel(const float* __restrict__ t,
                       const float* __restrict__ sumb,
                       const float* __restrict__ sqb,
                       const float* __restrict__ gamma,
                       const float* __restrict__ beta,
                       const float* __restrict__ addsrc,  // nullable
                       float* __restrict__ dst,
                       int totalVec)
{
    int i = blockIdx.x * blockDim.x + threadIdx.x;
    if (i >= totalVec) return;
    size_t base = (size_t)i * 4;
    int ff = (int)(base % NF);
    const float invB = 1.0f / (float)B_TOT;

    float4 tv = *(const float4*)(t + base);
    float4 sv = *(const float4*)(sumb + ff);
    float4 qv = *(const float4*)(sqb + ff);
    float4 gv = *(const float4*)(gamma + ff);
    float4 bv = *(const float4*)(beta + ff);

    float in[4]  = { tv.x, tv.y, tv.z, tv.w };
    float sm[4]  = { sv.x, sv.y, sv.z, sv.w };
    float sq[4]  = { qv.x, qv.y, qv.z, qv.w };
    float ga[4]  = { gv.x, gv.y, gv.z, gv.w };
    float be[4]  = { bv.x, bv.y, bv.z, bv.w };
    float out[4];

    #pragma unroll
    for (int k = 0; k < 4; ++k) {
        float mu  = sm[k] * invB;
        float var = sq[k] * invB - mu * mu;          // ddof=0 (jnp.var)
        float sc  = ga[k] * __frsqrt_rn(var + BN_EPS);
        float z   = (in[k] - mu) * sc + be[k];
        out[k] = 1.0f / (1.0f + __expf(-z));         // sigmoid
    }
    if (addsrc) {
        float4 av = *(const float4*)(addsrc + base);
        out[0] += av.x; out[1] += av.y; out[2] += av.z; out[3] += av.w;
    }
    float4 ov = { out[0], out[1], out[2], out[3] };
    *(float4*)(dst + base) = ov;
}

__global__ void zero_kernel(float* __restrict__ p, int n)
{
    int i = blockIdx.x * blockDim.x + threadIdx.x;
    if (i < n) p[i] = 0.0f;
}

// ---------------------------------------------------------------------------
// Inputs (setup_inputs order): x[B,48,64], W[11,64,64], att[11,48,48],
// bias[11,64], gamma[11,3072], beta[11,3072].  Output: y11 + x, f32.
// Workspace: y1..y5 (kept for DenseNet residuals) + hbuf + tbuf + stats.
// ---------------------------------------------------------------------------
extern "C" void kernel_launch(void* const* d_in, const int* in_sizes, int n_in,
                              void* d_out, int out_size, void* d_ws, size_t ws_size,
                              hipStream_t stream)
{
    const float* x     = (const float*)d_in[0];
    const float* W     = (const float*)d_in[1];
    const float* att   = (const float*)d_in[2];
    const float* bias  = (const float*)d_in[3];
    const float* gamma = (const float*)d_in[4];
    const float* beta  = (const float*)d_in[5];
    float* out = (float*)d_out;

    const size_t ACT = (size_t)B_TOT * N_NODE * F_FEAT;   // 25,165,824 floats
    float* ws = (float*)d_ws;
    float* ybuf[6];
    for (int k = 1; k <= 5; ++k) ybuf[k] = ws + (size_t)(k - 1) * ACT;
    float* hbuf = ws + 5 * ACT;
    float* tbuf = ws + 6 * ACT;
    float* sumb = ws + 7 * ACT;                 // [11][3072]
    float* sqb  = sumb + (size_t)L_LAYER * NF;  // [11][3072]

    const int statN = 2 * L_LAYER * NF;
    zero_kernel<<<(statN + 255) / 256, 256, 0, stream>>>(sumb, statN);

    const int totalVec = (int)(ACT / 4);        // 6,291,456
    const int bnBlocks = totalVec / 256;        // 24,576 (exact)

    const float* h = x;
    for (int i = 0; i < L_LAYER; ++i) {
        gcn_gemm_kernel<<<GEMM_WGS, 128, 0, stream>>>(
            h,
            W   + (size_t)i * 64 * 64,
            att + (size_t)i * 48 * 48,
            bias + (size_t)i * 64,
            tbuf,
            sumb + (size_t)i * NF,
            sqb  + (size_t)i * NF);

        const float* adds; float* dst;
        if (i < 5)       { adds = nullptr;        dst = ybuf[i + 1]; } // y1..y5 kept
        else if (i < 10) { adds = ybuf[10 - i];   dst = hbuf;        } // h_{i+2}=y_{i+1}+y_{10-i}
        else             { adds = x;              dst = out;         } // final: y11 + x

        bn_sigmoid_kernel<<<bnBlocks, 256, 0, stream>>>(
            tbuf,
            sumb + (size_t)i * NF, sqb + (size_t)i * NF,
            gamma + (size_t)i * NF, beta + (size_t)i * NF,
            adds, dst, totalVec);

        h = dst;
    }
}
